// CAM_Module_71073118814824
// MI455X (gfx1250) — matrix operand order repaired and checked by reading, not compiled, not run on hardware
//
#include <hip/hip_runtime.h>
#include <hip/hip_bf16.h>
#include <stdint.h>

#define B_DIM 32
#define C_DIM 512
#define N_DIM 4096
#define PITCH 72   // 64 bf16 payload (128B) + 16B pad per row -> 144B row stride

typedef __bf16 bf16_t;
typedef bf16_t v16bf  __attribute__((ext_vector_type(16)));
typedef bf16_t bf16x8 __attribute__((ext_vector_type(8)));
typedef float  v8f    __attribute__((ext_vector_type(8)));
typedef unsigned int u32x4 __attribute__((ext_vector_type(4)));
typedef int          i32x8 __attribute__((ext_vector_type(8)));
typedef int          i32x4 __attribute__((ext_vector_type(4)));

__device__ __forceinline__ unsigned short f2bf(float f) {
  unsigned int u = __float_as_uint(f);
  u += 0x7FFFu + ((u >> 16) & 1u);           // round-to-nearest-even
  return (unsigned short)(u >> 16);
}

__device__ __forceinline__ v16bf ld_frag(const unsigned short* p0, const unsigned short* p1) {
  bf16x8 lo = *reinterpret_cast<const bf16x8*>(p0);
  bf16x8 hi = *reinterpret_cast<const bf16x8*>(p1);
  return __builtin_shufflevector(lo, hi, 0,1,2,3,4,5,6,7,8,9,10,11,12,13,14,15);
}

// Shared 64x64-tile x K=64-stage WMMA compute: 2 chunks x (2x2) wmma per wave.
__device__ __forceinline__ void gemm_stage(const unsigned short* Ap, const unsigned short* Bp,
                                           int wr, int wc, int m, int kh, v8f acc[2][2]) {
#pragma unroll
  for (int c = 0; c < 2; ++c) {
    v16bf af[2], bfv[2];
#pragma unroll
    for (int mi = 0; mi < 2; ++mi) {
      int ar = wr * 32 + mi * 16 + m;
      af[mi] = ld_frag(&Ap[ar * PITCH + c * 32 + kh * 8],
                       &Ap[ar * PITCH + c * 32 + 16 + kh * 8]);
    }
#pragma unroll
    for (int ni = 0; ni < 2; ++ni) {
      int br = wc * 32 + ni * 16 + m;          // lane m is the N index for B
      bfv[ni] = ld_frag(&Bp[br * PITCH + c * 32 + kh * 8],
                        &Bp[br * PITCH + c * 32 + 16 + kh * 8]);
    }
#pragma unroll
    for (int mi = 0; mi < 2; ++mi)
#pragma unroll
      for (int ni = 0; ni < 2; ++ni)
        acc[mi][ni] = __builtin_amdgcn_wmma_f32_16x16x32_bf16(
            false, af[mi], false, bfv[ni], (short)0, acc[mi][ni], false, false);
  }
}

// Tensor Data Mover: load a 32-row x 64-bf16 tile (row stride N_DIM elements) into
// LDS at lds_addr with 16B padding per 128B row (-> 144B LDS pitch).
__device__ __forceinline__ void tdm_load_tile(unsigned lds_addr, const unsigned short* gptr) {
  unsigned long long ga = (unsigned long long)(uintptr_t)gptr;
  u32x4 g0;
  g0.x = 1u;                                            // count=1 (valid), user mode
  g0.y = lds_addr;                                      // lds_addr [63:32]
  g0.z = (unsigned)(ga & 0xffffffffull);                // global_addr [95:64]
  g0.w = (unsigned)((ga >> 32) & 0x1ffffffull) | (2u << 30);  // addr[56:32] | type=2
  i32x8 g1;
  g1[0] = (int)((1u << 16)    // data_size = 1 -> 2 bytes
              | (1u << 20)    // pad_enable
              | (4u << 22)    // pad_interval = 32 dwords (128B)
              | (3u << 25));  // pad_amount   = 4 dwords (16B)
  g1[1] = (int)((unsigned)N_DIM << 16);   // tensor_dim0[15:0]
  g1[2] = (int)((unsigned)C_DIM << 16);   // tensor_dim1[15:0]
  g1[3] = (int)(64u << 16);               // tile_dim0 = 64
  g1[4] = 32;                             // tile_dim1 = 32
  g1[5] = N_DIM;                          // tensor_dim0_stride[31:0]
  g1[6] = 0;
  g1[7] = 0;
  i32x4 z4 = {0, 0, 0, 0};
#if __clang_major__ >= 23
  i32x8 z8 = {0, 0, 0, 0, 0, 0, 0, 0};
  __builtin_amdgcn_tensor_load_to_lds(g0, g1, z4, z4, z8, 0);
#else
  __builtin_amdgcn_tensor_load_to_lds(g0, g1, z4, z4, 0);
#endif
}

// One TDM descriptor per wave: each wave DMA's half of one 64x64 tile.
__device__ __forceinline__ void issue_energy_stage(int w, unsigned ldsA, unsigned ldsB,
                                                   const unsigned short* X,
                                                   int rowA, int rowB, int k0) {
  if (w == 0)      tdm_load_tile(ldsA,            X + (size_t)(rowA +  0) * N_DIM + k0);
  else if (w == 1) tdm_load_tile(ldsA + 32 * 144, X + (size_t)(rowA + 32) * N_DIM + k0);
  else if (w == 2) tdm_load_tile(ldsB,            X + (size_t)(rowB +  0) * N_DIM + k0);
  else             tdm_load_tile(ldsB + 32 * 144, X + (size_t)(rowB + 32) * N_DIM + k0);
}

// ---------------- Kernel 1: f32 -> bf16 convert ----------------
__global__ void k_cvt_bf16(const float* __restrict__ x, unsigned short* __restrict__ xb,
                           long long n4) {
  long long i = (long long)blockIdx.x * blockDim.x + threadIdx.x;
  long long stride = (long long)gridDim.x * blockDim.x;
  const float4* src = reinterpret_cast<const float4*>(x);
  uint2* dst = reinterpret_cast<uint2*>(xb);
  for (; i < n4; i += stride) {
    float4 v = src[i];
    union { unsigned short s[4]; uint2 u; } pk;
    pk.s[0] = f2bf(v.x); pk.s[1] = f2bf(v.y);
    pk.s[2] = f2bf(v.z); pk.s[3] = f2bf(v.w);
    dst[i] = pk.u;
  }
}

// -------- Kernel 2: E = X * X^T  (double-buffered TDM pipeline, bf16 WMMA) --------
__global__ void __launch_bounds__(128) k_energy(const unsigned short* __restrict__ xb,
                                                float* __restrict__ energy) {
  __shared__ __align__(16) unsigned short As[2][64 * PITCH];
  __shared__ __align__(16) unsigned short Bs[2][64 * PITCH];

  const int b  = blockIdx.x >> 6;
  const int t  = blockIdx.x & 63;
  const int ti = t >> 3, tj = t & 7;               // 64x64 tile of 512x512 E
  const int tid  = threadIdx.x;
  const int lane = tid & 31;
  const int w  = tid >> 5, wr = w >> 1, wc = w & 1; // 2x2 waves -> 32x32 subtile each
  const int m  = lane & 15, kh = lane >> 4;
  const int rowA = ti * 64, rowB = tj * 64;

  const unsigned short* X = xb + (size_t)b * C_DIM * N_DIM;
  const unsigned ldsA0 = (unsigned)(uintptr_t)&As[0][0];
  const unsigned ldsA1 = (unsigned)(uintptr_t)&As[1][0];
  const unsigned ldsB0 = (unsigned)(uintptr_t)&Bs[0][0];
  const unsigned ldsB1 = (unsigned)(uintptr_t)&Bs[1][0];

  v8f acc[2][2] = {};

  // Prime the pipeline: stage 0 DMA into buffer 0.
  issue_energy_stage(w, ldsA0, ldsB0, X, rowA, rowB, 0);

#pragma unroll 1
  for (int k0 = 0; k0 < N_DIM; k0 += 128) {
    // ---- stage k0 in buffer 0 ----
    __builtin_amdgcn_s_wait_tensorcnt(0);   // own DMA done
    __syncthreads();                        // all tiles visible; buf1 readers drained
    issue_energy_stage(w, ldsA1, ldsB1, X, rowA, rowB, k0 + 64);  // DMA overlaps WMMA
    gemm_stage(As[0], Bs[0], wr, wc, m, kh, acc);

    // ---- stage k0+64 in buffer 1 ----
    __builtin_amdgcn_s_wait_tensorcnt(0);
    __syncthreads();
    if (k0 + 128 < N_DIM)
      issue_energy_stage(w, ldsA0, ldsB0, X, rowA, rowB, k0 + 128);
    gemm_stage(As[1], Bs[1], wr, wc, m, kh, acc);
  }

  float* Eb = energy + (size_t)b * C_DIM * C_DIM;
  const int hi = lane >> 4, n = lane & 15;
#pragma unroll
  for (int mi = 0; mi < 2; ++mi)
#pragma unroll
    for (int ni = 0; ni < 2; ++ni)
#pragma unroll
      for (int r = 0; r < 8; ++r) {
        int row = rowA + wr * 32 + mi * 16 + r + hi * 8;
        int col = rowB + wc * 32 + ni * 16 + n;
        Eb[(size_t)row * C_DIM + col] = acc[mi][ni][r];
      }
}

// ---------------- Kernel 3: inverted softmax per row (rowmax cancels; use rowmin) ----
__global__ void __launch_bounds__(128) k_softmax(float* __restrict__ energy) {
  __shared__ float red[128];
  float* row = energy + (size_t)blockIdx.x * C_DIM;
  const int tid = threadIdx.x;

  float v[4];
  float mn = 3.4e38f;
#pragma unroll
  for (int e = 0; e < 4; ++e) { v[e] = row[tid + e * 128]; mn = fminf(mn, v[e]); }
  red[tid] = mn; __syncthreads();
  for (int s = 64; s > 0; s >>= 1) {
    if (tid < s) red[tid] = fminf(red[tid], red[tid + s]);
    __syncthreads();
  }
  const float rmin = red[0];
  __syncthreads();

  float ev[4]; float sum = 0.f;
#pragma unroll
  for (int e = 0; e < 4; ++e) { ev[e] = __expf(rmin - v[e]); sum += ev[e]; }
  red[tid] = sum; __syncthreads();
  for (int s = 64; s > 0; s >>= 1) {
    if (tid < s) red[tid] += red[tid + s];
    __syncthreads();
  }
  const float inv = 1.0f / red[0];
#pragma unroll
  for (int e = 0; e < 4; ++e) row[tid + e * 128] = ev[e] * inv;
}

// ---------------- Kernel 4: out = gamma * (A @ X) + x (software-pipelined) ----------
__device__ __forceinline__ void kout_load_stage(const float* __restrict__ Abase,
                                                const unsigned short* __restrict__ Bbase,
                                                float4 fa[8], uint4 qb[4]) {
  const float4* sa = reinterpret_cast<const float4*>(Abase);
  const uint4*  sb = reinterpret_cast<const uint4*>(Bbase);
#pragma unroll
  for (int q = 0; q < 8; ++q) fa[q] = sa[q];
#pragma unroll
  for (int q = 0; q < 4; ++q) qb[q] = sb[q];
}

__device__ __forceinline__ void kout_store_stage(unsigned short* Ap, unsigned short* Bp,
                                                 int lr, int lh,
                                                 const float4 fa[8], const uint4 qb[4]) {
  union { unsigned short s[32]; uint4 q[4]; } pk;
#pragma unroll
  for (int q = 0; q < 8; ++q) {
    pk.s[q * 4 + 0] = f2bf(fa[q].x); pk.s[q * 4 + 1] = f2bf(fa[q].y);
    pk.s[q * 4 + 2] = f2bf(fa[q].z); pk.s[q * 4 + 3] = f2bf(fa[q].w);
  }
#pragma unroll
  for (int q = 0; q < 4; ++q)
    *reinterpret_cast<uint4*>(&Ap[lr * PITCH + lh * 32 + q * 8]) = pk.q[q];
  union { unsigned short s[32]; uint4 q[4]; } bb;
  bb.q[0] = qb[0]; bb.q[1] = qb[1]; bb.q[2] = qb[2]; bb.q[3] = qb[3];
#pragma unroll
  for (int e = 0; e < 32; ++e)        // transpose scatter: Bs[n][j]
    Bp[(lh * 32 + e) * PITCH + lr] = bb.s[e];
}

__global__ void __launch_bounds__(128) k_out(const unsigned short* __restrict__ xb,
                                             const float* __restrict__ attn,
                                             const float* __restrict__ x,
                                             const float* __restrict__ gamma,
                                             float* __restrict__ out) {
  __shared__ __align__(16) unsigned short As[2][64 * PITCH];
  __shared__ __align__(16) unsigned short Bs[2][64 * PITCH];

  const int b  = blockIdx.x >> 9;
  const int t  = blockIdx.x & 511;
  const int ti = t >> 6;                 // i-tile (0..7)  -> rows of out
  const int tn = t & 63;                 // n-tile (0..63) -> cols of out
  const int i0 = ti * 64, n0 = tn * 64;
  const int tid  = threadIdx.x;
  const int lane = tid & 31;
  const int w  = tid >> 5, wr = w >> 1, wc = w & 1;
  const int m  = lane & 15, kh = lane >> 4;
  const int lr = tid >> 1, lh = tid & 1;   // loader: row, 32-elem half

  const float*          Ab = attn + (size_t)b * C_DIM * C_DIM + (size_t)(i0 + lr) * C_DIM + lh * 32;
  const unsigned short* Xb = xb   + (size_t)b * C_DIM * N_DIM + (size_t)lr * N_DIM + n0 + lh * 32;

  v8f acc[2][2] = {};
  float4 fa[8]; uint4 qb[4];

  // Prime: stage 0 through registers into buffer 0, then start stage-1 loads.
  kout_load_stage(Ab + 0, Xb + (size_t)0 * N_DIM, fa, qb);
  kout_store_stage(As[0], Bs[0], lr, lh, fa, qb);
  kout_load_stage(Ab + 64, Xb + (size_t)64 * N_DIM, fa, qb);

  const int NS = C_DIM / 64;   // 8 stages
#pragma unroll 1
  for (int k = 0; k < NS; k += 2) {
    __syncthreads();                                   // buf0(stage k) stored; buf1 drained
    kout_store_stage(As[1], Bs[1], lr, lh, fa, qb);    // stage k+1 -> buf1
    if (k + 2 < NS) {
      kout_load_stage(Ab + (k + 2) * 64, Xb + (size_t)(k + 2) * 64 * N_DIM, fa, qb);
      if (k + 4 < NS) {                                // warm L2 two stages ahead
        __builtin_prefetch(Ab + (k + 4) * 64, 0, 1);
        __builtin_prefetch(Xb + (size_t)(k + 4) * 64 * N_DIM, 0, 1);
      }
    }
    gemm_stage(As[0], Bs[0], wr, wc, m, kh, acc);      // stage k

    __syncthreads();                                   // buf1(stage k+1) stored; buf0 drained
    if (k + 2 < NS) {
      kout_store_stage(As[0], Bs[0], lr, lh, fa, qb);  // stage k+2 -> buf0
      if (k + 3 < NS)
        kout_load_stage(Ab + (k + 3) * 64, Xb + (size_t)(k + 3) * 64 * N_DIM, fa, qb);
    }
    gemm_stage(As[1], Bs[1], wr, wc, m, kh, acc);      // stage k+1
  }

  const float g = gamma[0];
  const int hi = lane >> 4, n = lane & 15;
#pragma unroll
  for (int mi = 0; mi < 2; ++mi)
#pragma unroll
    for (int ni = 0; ni < 2; ++ni)
#pragma unroll
      for (int r = 0; r < 8; ++r) {
        int row = i0 + wr * 32 + mi * 16 + r + hi * 8;
        int col = n0 + wc * 32 + ni * 16 + n;
        size_t idx = ((size_t)b * C_DIM + row) * N_DIM + col;
        out[idx] = g * acc[mi][ni][r] + x[idx];
      }
}

extern "C" void kernel_launch(void* const* d_in, const int* in_sizes, int n_in,
                              void* d_out, int out_size, void* d_ws, size_t ws_size,
                              hipStream_t stream) {
  const float* x     = (const float*)d_in[0];
  const float* gamma = (const float*)d_in[1];
  float* out = (float*)d_out;

  // Workspace layout: [0, 128MB) bf16 copy of x; [128MB, 160MB) f32 energy/attention
  unsigned short* xb = (unsigned short*)d_ws;
  float* energy = (float*)((char*)d_ws + (size_t)B_DIM * C_DIM * N_DIM * 2);

  long long n4 = (long long)B_DIM * C_DIM * N_DIM / 4;
  k_cvt_bf16<<<8192, 256, 0, stream>>>(x, xb, n4);
  k_energy <<<B_DIM * 64, 128, 0, stream>>>(xb, energy);
  k_softmax<<<B_DIM * C_DIM, 128, 0, stream>>>(energy);
  k_out    <<<B_DIM * 8 * 64, 128, 0, stream>>>(xb, energy, x, gamma, out);
}
